// Attention_89902255439889
// MI455X (gfx1250) — compile-verified, matrix operand order checked by reading, not run on hardware
//
#include <hip/hip_runtime.h>

#define EDIM 1024
#define HDIM 16
#define DDIM 64
#define BB 4
#define SS 2048
#define BSTOK (BB * SS)   // 8192 tokens

typedef __attribute__((ext_vector_type(16))) __bf16          bf16x16;
typedef __attribute__((ext_vector_type(16))) unsigned short  u16x16;
typedef __attribute__((ext_vector_type(8)))  unsigned short  u16x8;
typedef __attribute__((ext_vector_type(8)))  float           v8f;
typedef int v4i __attribute__((vector_size(16)));            // 128-bit payload

#if __has_builtin(__builtin_amdgcn_global_load_async_to_lds_b128)
#define USE_ASYNC 1
#else
#define USE_ASYNC 0
#endif

// ---------------------------------------------------------------- helpers ---

__device__ __forceinline__ unsigned short f2bf(float f) {
    unsigned int u = __float_as_uint(f);
    u += 0x7fffu + ((u >> 16) & 1u);          // round-to-nearest-even
    return (unsigned short)(u >> 16);
}

__device__ __forceinline__ v8f wmma_bf16(bf16x16 a, bf16x16 b, v8f c) {
    // D = A(16x32 bf16) * B(32x16 bf16) + C(16x16 f32)
    return __builtin_amdgcn_wmma_f32_16x16x32_bf16(
        /*neg_a=*/false, a, /*neg_b=*/false, b,
        /*c_mod=*/(short)0, c, /*reuse_a=*/false, /*reuse_b=*/false);
}

// 16-byte global -> LDS copy; async (ASYNCcnt) when the builtin exists.
__device__ __forceinline__ void cp_g2l_b128(const unsigned short* g,
                                            unsigned short* l) {
#if USE_ASYNC
    unsigned long long ga = (unsigned long long)g;
    unsigned int la = (unsigned int)(unsigned long long)l;  // low 32b = LDS off
    __builtin_amdgcn_global_load_async_to_lds_b128(
        (__attribute__((address_space(1))) v4i*)ga,
        (__attribute__((address_space(3))) v4i*)la, 0, 0);
#else
    *(u16x8*)l = *(const u16x8*)g;
#endif
}

__device__ __forceinline__ void wait_async_done() {
#if USE_ASYNC
#if __has_builtin(__builtin_amdgcn_s_wait_asynccnt)
    __builtin_amdgcn_s_wait_asynccnt(0);
#else
    asm volatile("s_wait_asynccnt 0x0" ::: "memory");
#endif
#endif
}

// Load one 16x32 bf16 operand fragment from a row-major ushort(bf16) source.
// Per ISA 16-bit A/B layout: lane r=(lane&15) holds row r; half=(lane>>4)
// selects K chunks {half*8 .. half*8+7} and {16+half*8 .. 16+half*8+7}.
__device__ __forceinline__ bf16x16 load_frag(const unsigned short* base, int ld) {
    const int lane = threadIdx.x & 31;
    const int r    = lane & 15;
    const int hf   = lane >> 4;
    const unsigned short* p = base + (size_t)r * ld + hf * 8;
    u16x8 c0 = *(const u16x8*)(p);
    u16x8 c1 = *(const u16x8*)(p + 16);
    u16x16 v;
#pragma unroll
    for (int i = 0; i < 8; ++i) { v[i] = c0[i]; v[8 + i] = c1[i]; }
    return __builtin_bit_cast(bf16x16, v);
}

// ------------------------------------------------------------ cvt kernel ---

__global__ void k_cvt_bf16(const float* __restrict__ src,
                           unsigned short* __restrict__ dst, int n) {
    int i = blockIdx.x * blockDim.x + threadIdx.x;
    int stride = gridDim.x * blockDim.x;
    for (; i < n; i += stride) dst[i] = f2bf(src[i]);
}

// ---------------------------------------------- QKV projection (bf16 WMMA) ---
// grid = (BSTOK/128, 3*H). Block: 256 thr = 8 waves; wave w owns 16 token rows.
// Computes Y = X * W^T + b for one head's 64 columns, writes [B,H,S,D] bf16.

__global__ __launch_bounds__(256) void k_qkv(
    const unsigned short* __restrict__ xb,
    const unsigned short* __restrict__ wqb,
    const unsigned short* __restrict__ wkb,
    const unsigned short* __restrict__ wvb,
    const float* __restrict__ bq, const float* __restrict__ bk,
    const float* __restrict__ bv,
    unsigned short* __restrict__ Qb, unsigned short* __restrict__ Kb,
    unsigned short* __restrict__ Vb) {
    __shared__ unsigned short lds_w[64 * 32];   // W tile [64 n x 32 k]

    const int tid  = threadIdx.x;
    const int lane = tid & 31;
    const int w    = tid >> 5;
    const int n    = lane & 15;
    const int hf   = lane >> 4;
    const int t0   = blockIdx.x * 128;
    const int proj = blockIdx.y >> 4;
    const int head = blockIdx.y & 15;

    const unsigned short* W = (proj == 0) ? wqb : (proj == 1) ? wkb : wvb;
    const float* bias       = (proj == 0) ? bq  : (proj == 1) ? bk  : bv;
    unsigned short* dst     = (proj == 0) ? Qb  : (proj == 1) ? Kb  : Vb;

    v8f acc[4] = {};
    const int srow = tid >> 2;          // 0..63
    const int sc   = (tid & 3) * 8;     // 0,8,16,24

    for (int k0 = 0; k0 < EDIM; k0 += 32) {
        // stage W tile (shared by all 8 waves)
        cp_g2l_b128(&W[(size_t)(head * 64 + srow) * EDIM + k0 + sc],
                    &lds_w[srow * 32 + sc]);
        if (k0 + 32 < EDIM)
            __builtin_prefetch(&W[(size_t)(head * 64 + srow) * EDIM + k0 + 32 + sc],
                               0, 3);
        wait_async_done();
        __syncthreads();

        bf16x16 a = load_frag(&xb[(size_t)(t0 + w * 16) * EDIM + k0], EDIM);
#pragma unroll
        for (int nt = 0; nt < 4; ++nt) {
            bf16x16 b = load_frag(&lds_w[nt * 16 * 32], 32);
            acc[nt] = wmma_bf16(a, b, acc[nt]);
        }
        __syncthreads();
    }

#pragma unroll
    for (int nt = 0; nt < 4; ++nt) {
#pragma unroll
        for (int j = 0; j < 8; ++j) {
            int col = head * 64 + nt * 16 + n;
            int tok = t0 + w * 16 + j + 8 * hf;
            int bi  = tok >> 11;            // / S
            int si  = tok & (SS - 1);
            float v = acc[nt][j] + bias[col];
            dst[(((size_t)bi * HDIM + head) * SS + si) * DDIM + nt * 16 + n] =
                f2bf(v);
        }
    }
}

// ----------------------------------------- flash attention (bf16 WMMA) ------
// grid = (S/128, B*H). Wave w owns q rows [q0, q0+16); 64-key tiles; online
// softmax in fp32 via half-wave shuffle reductions (C-layout rows live in one
// 16-lane half).

__global__ __launch_bounds__(256) void k_attn(
    const unsigned short* __restrict__ Qb,
    const unsigned short* __restrict__ Kb,
    const unsigned short* __restrict__ Vb,
    unsigned short* __restrict__ ctx) {
    __shared__ unsigned short lds_k[64 * 64];        // K tile [key][d]   8 KB
    __shared__ unsigned short lds_vt[64 * 72];       // V^T [d][key] pad  9 KB
    __shared__ unsigned short lds_p[8][16 * 64];     // per-wave probs   16 KB

    const int tid  = threadIdx.x;
    const int lane = tid & 31;
    const int w    = tid >> 5;
    const int n    = lane & 15;
    const int hf   = lane >> 4;
    const int bh   = blockIdx.y;     // b*H + h
    const int bi   = bh >> 4;
    const int head = bh & 15;
    const int q0   = blockIdx.x * 128 + w * 16;

    const size_t headBase = (size_t)bh * SS * DDIM;

    bf16x16 qa0 = load_frag(&Qb[headBase + (size_t)q0 * DDIM + 0],  DDIM);
    bf16x16 qa1 = load_frag(&Qb[headBase + (size_t)q0 * DDIM + 32], DDIM);

    v8f acc[4] = {};
    float m[8], l[8];
#pragma unroll
    for (int j = 0; j < 8; ++j) { m[j] = -__builtin_inff(); l[j] = 0.0f; }

    const float scale = 0.125f;              // 1/sqrt(64)
    const int key0 = tid >> 3;               // 0..31
    const int key1 = key0 + 32;              // 32..63
    const int sc   = (tid & 7) * 8;          // d chunk

    for (int kt = 0; kt < SS; kt += 64) {
        // K tile: straight copy (async path when available)
        cp_g2l_b128(&Kb[headBase + (size_t)(kt + key0) * DDIM + sc],
                    &lds_k[key0 * 64 + sc]);
        cp_g2l_b128(&Kb[headBase + (size_t)(kt + key1) * DDIM + sc],
                    &lds_k[key1 * 64 + sc]);
        // V tile: register transpose -> lds_vt[d][key], ld = 72
        u16x8 vv0 = *(const u16x8*)&Vb[headBase + (size_t)(kt + key0) * DDIM + sc];
        u16x8 vv1 = *(const u16x8*)&Vb[headBase + (size_t)(kt + key1) * DDIM + sc];
#pragma unroll
        for (int e = 0; e < 8; ++e) {
            lds_vt[(sc + e) * 72 + key0] = vv0[e];
            lds_vt[(sc + e) * 72 + key1] = vv1[e];
        }
        wait_async_done();
        __syncthreads();

        // scores: Q(16x64) @ K^T(64x64) as four 16-key N tiles, K-depth 64
        v8f s[4];
#pragma unroll
        for (int t = 0; t < 4; ++t) {
            v8f z = {};
            bf16x16 b0 = load_frag(&lds_k[t * 16 * 64 + 0],  64);
            z = wmma_bf16(qa0, b0, z);
            bf16x16 b1 = load_frag(&lds_k[t * 16 * 64 + 32], 64);
            s[t] = wmma_bf16(qa1, b1, z);
        }

        // online softmax; row = j + 8*hf
        unsigned short* pw = &lds_p[w][0];
#pragma unroll
        for (int j = 0; j < 8; ++j) {
            float v0 = s[0][j] * scale;
            float v1 = s[1][j] * scale;
            float v2 = s[2][j] * scale;
            float v3 = s[3][j] * scale;
            float rmax = fmaxf(fmaxf(v0, v1), fmaxf(v2, v3));
            rmax = fmaxf(rmax, __shfl_xor(rmax, 1, 32));
            rmax = fmaxf(rmax, __shfl_xor(rmax, 2, 32));
            rmax = fmaxf(rmax, __shfl_xor(rmax, 4, 32));
            rmax = fmaxf(rmax, __shfl_xor(rmax, 8, 32));
            float mn   = fmaxf(m[j], rmax);
            float corr = __expf(m[j] - mn);
            m[j] = mn;
            float p0 = __expf(v0 - mn);
            float p1 = __expf(v1 - mn);
            float p2 = __expf(v2 - mn);
            float p3 = __expf(v3 - mn);
            float rs = (p0 + p1) + (p2 + p3);
            rs += __shfl_xor(rs, 1, 32);
            rs += __shfl_xor(rs, 2, 32);
            rs += __shfl_xor(rs, 4, 32);
            rs += __shfl_xor(rs, 8, 32);
            l[j] = l[j] * corr + rs;
#pragma unroll
            for (int t = 0; t < 4; ++t) acc[t][j] *= corr;
            int row = j + 8 * hf;
            pw[row * 64 +  0 + n] = f2bf(p0);   // re-layout C -> A via LDS
            pw[row * 64 + 16 + n] = f2bf(p1);
            pw[row * 64 + 32 + n] = f2bf(p2);
            pw[row * 64 + 48 + n] = f2bf(p3);
        }

        // ctx += P(16x64) @ V(64x64); same-wave LDS RAW is in-order (DScnt)
        bf16x16 pa0 = load_frag(pw + 0,  64);
        bf16x16 pa1 = load_frag(pw + 32, 64);
#pragma unroll
        for (int t = 0; t < 4; ++t) {
            bf16x16 bv0 = load_frag(&lds_vt[t * 16 * 72 + 0],  72);
            acc[t] = wmma_bf16(pa0, bv0, acc[t]);
            bf16x16 bv1 = load_frag(&lds_vt[t * 16 * 72 + 32], 72);
            acc[t] = wmma_bf16(pa1, bv1, acc[t]);
        }
        __syncthreads();
    }

    // normalize by l and write ctx as [B,S,H,D] (== [token, E]) bf16
#pragma unroll
    for (int j = 0; j < 8; ++j) {
        float inv = 1.0f / l[j];
        int si = q0 + j + 8 * hf;
        size_t tok = (size_t)bi * SS + si;
#pragma unroll
        for (int t = 0; t < 4; ++t) {
            float v = acc[t][j] * inv;
            ctx[(tok * HDIM + head) * DDIM + t * 16 + n] = f2bf(v);
        }
    }
}

// -------------------- output projection + bias + residual + LayerNorm -------
// grid = BSTOK/16. Wave w covers columns [w*128, w*128+128); whole row in block.

__global__ __launch_bounds__(256) void k_oproj_ln(
    const unsigned short* __restrict__ ctx,
    const unsigned short* __restrict__ wob,
    const float* __restrict__ bo,
    const float* __restrict__ x,
    const float* __restrict__ ln_g,
    const float* __restrict__ ln_b,
    float* __restrict__ out) {
    __shared__ unsigned short lds_a[16 * 32];
    __shared__ float lds_sum[16];
    __shared__ float lds_sq[16];

    const int tid  = threadIdx.x;
    const int lane = tid & 31;
    const int w    = tid >> 5;
    const int n    = lane & 15;
    const int hf   = lane >> 4;
    const int t0   = blockIdx.x * 16;

    v8f acc[8] = {};
    for (int k0 = 0; k0 < EDIM; k0 += 32) {
        if (tid < 64) {
            int r = tid >> 2, c = (tid & 3) * 8;
            cp_g2l_b128(&ctx[(size_t)(t0 + r) * EDIM + k0 + c],
                        &lds_a[r * 32 + c]);
        }
        wait_async_done();
        __syncthreads();
        bf16x16 a = load_frag(lds_a, 32);
#pragma unroll
        for (int nt = 0; nt < 8; ++nt) {
            bf16x16 b =
                load_frag(&wob[(size_t)(w * 128 + nt * 16) * EDIM + k0], EDIM);
            acc[nt] = wmma_bf16(a, b, acc[nt]);
        }
        __syncthreads();
    }

    if (tid < 16) { lds_sum[tid] = 0.0f; lds_sq[tid] = 0.0f; }
    __syncthreads();

    // y = gemm + bias + residual; accumulate per-row mean / meansq
#pragma unroll
    for (int j = 0; j < 8; ++j) {
        int row = j + 8 * hf;
        size_t tok = (size_t)(t0 + row);
        float s = 0.0f, sq = 0.0f;
#pragma unroll
        for (int nt = 0; nt < 8; ++nt) {
            int col = w * 128 + nt * 16 + n;
            float y = acc[nt][j] + bo[col] + x[tok * EDIM + col];
            acc[nt][j] = y;
            s += y; sq += y * y;
        }
        s += __shfl_xor(s, 1, 32);  sq += __shfl_xor(sq, 1, 32);
        s += __shfl_xor(s, 2, 32);  sq += __shfl_xor(sq, 2, 32);
        s += __shfl_xor(s, 4, 32);  sq += __shfl_xor(sq, 4, 32);
        s += __shfl_xor(s, 8, 32);  sq += __shfl_xor(sq, 8, 32);
        if (n == 0) {
            atomicAdd(&lds_sum[row], s);
            atomicAdd(&lds_sq[row], sq);
        }
    }
    __syncthreads();

#pragma unroll
    for (int j = 0; j < 8; ++j) {
        int row = j + 8 * hf;
        size_t tok = (size_t)(t0 + row);
        float mu   = lds_sum[row] * (1.0f / EDIM);
        float var  = lds_sq[row] * (1.0f / EDIM) - mu * mu;
        float rstd = rsqrtf(var + 1e-5f);
#pragma unroll
        for (int nt = 0; nt < 8; ++nt) {
            int col = w * 128 + nt * 16 + n;
            out[tok * EDIM + col] =
                (acc[nt][j] - mu) * rstd * ln_g[col] + ln_b[col];
        }
    }
}

// ------------------------------------------------------------------ launch ---

extern "C" void kernel_launch(void* const* d_in, const int* in_sizes, int n_in,
                              void* d_out, int out_size, void* d_ws,
                              size_t ws_size, hipStream_t stream) {
    (void)in_sizes; (void)n_in; (void)out_size; (void)ws_size;
    const float* x  = (const float*)d_in[0];
    const float* Wq = (const float*)d_in[1];
    const float* bq = (const float*)d_in[2];
    const float* Wk = (const float*)d_in[3];
    const float* bk = (const float*)d_in[4];
    const float* Wv = (const float*)d_in[5];
    const float* bv = (const float*)d_in[6];
    const float* Wo = (const float*)d_in[7];
    const float* bo = (const float*)d_in[8];
    const float* lg = (const float*)d_in[9];
    const float* lb = (const float*)d_in[10];
    float* out = (float*)d_out;

    unsigned short* ws  = (unsigned short*)d_ws;
    unsigned short* xb  = ws;  ws += (size_t)BSTOK * EDIM;
    unsigned short* wqb = ws;  ws += (size_t)EDIM * EDIM;
    unsigned short* wkb = ws;  ws += (size_t)EDIM * EDIM;
    unsigned short* wvb = ws;  ws += (size_t)EDIM * EDIM;
    unsigned short* wob = ws;  ws += (size_t)EDIM * EDIM;
    unsigned short* Qb  = ws;  ws += (size_t)BSTOK * EDIM;
    unsigned short* Kb  = ws;  ws += (size_t)BSTOK * EDIM;
    unsigned short* Vb  = ws;  ws += (size_t)BSTOK * EDIM;
    unsigned short* ctx = ws;

    hipLaunchKernelGGL(k_cvt_bf16, dim3(2048), dim3(256), 0, stream, x,  xb,
                       BSTOK * EDIM);
    hipLaunchKernelGGL(k_cvt_bf16, dim3(512), dim3(256), 0, stream, Wq, wqb,
                       EDIM * EDIM);
    hipLaunchKernelGGL(k_cvt_bf16, dim3(512), dim3(256), 0, stream, Wk, wkb,
                       EDIM * EDIM);
    hipLaunchKernelGGL(k_cvt_bf16, dim3(512), dim3(256), 0, stream, Wv, wvb,
                       EDIM * EDIM);
    hipLaunchKernelGGL(k_cvt_bf16, dim3(512), dim3(256), 0, stream, Wo, wob,
                       EDIM * EDIM);

    hipLaunchKernelGGL(k_qkv, dim3(BSTOK / 128, 3 * HDIM), dim3(256), 0, stream,
                       xb, wqb, wkb, wvb, bq, bk, bv, Qb, Kb, Vb);
    hipLaunchKernelGGL(k_attn, dim3(SS / 128, BB * HDIM), dim3(256), 0, stream,
                       Qb, Kb, Vb, ctx);
    hipLaunchKernelGGL(k_oproj_ln, dim3(BSTOK / 16), dim3(256), 0, stream, ctx,
                       wob, bo, x, lg, lb, out);
}
